// DNCMemory_36507222016381
// MI455X (gfx1250) — matrix operand order repaired and checked by reading, not compiled, not run on hardware
//
#include <hip/hip_runtime.h>
#include <hip/hip_bf16.h>
#include <math.h>

typedef __attribute__((ext_vector_type(16))) __bf16 v16bf;
typedef __attribute__((ext_vector_type(8)))  float  v8f;

// ---------- dims ----------
#define KN 8192
#define KB 32
#define KH 256

// ---------- helpers ----------
__device__ __forceinline__ float sigf(float x) { return 1.0f / (1.0f + expf(-x)); }
__device__ __forceinline__ float softplusf(float x) { return (x > 20.0f) ? x : log1pf(expf(x)); }

struct U32x8 { uint4 a, b; };

// Load a v16bf fragment whose two 8-element halves are contiguous bf16 runs.
__device__ __forceinline__ v16bf ld_frag_bf(const __bf16* p0, const __bf16* p1) {
  U32x8 u;
  u.a = *reinterpret_cast<const uint4*>(p0);
  u.b = *reinterpret_cast<const uint4*>(p1);
  return __builtin_bit_cast(v16bf, u);
}
// Load two contiguous 8-float runs and convert to a v16bf fragment (native cvt).
__device__ __forceinline__ v16bf cvt_frag_f32(const float* p0, const float* p1) {
  float4 x0 = reinterpret_cast<const float4*>(p0)[0];
  float4 x1 = reinterpret_cast<const float4*>(p0)[1];
  float4 y0 = reinterpret_cast<const float4*>(p1)[0];
  float4 y1 = reinterpret_cast<const float4*>(p1)[1];
  v16bf r;
  r[0] = (__bf16)x0.x;  r[1] = (__bf16)x0.y;  r[2] = (__bf16)x0.z;  r[3] = (__bf16)x0.w;
  r[4] = (__bf16)x1.x;  r[5] = (__bf16)x1.y;  r[6] = (__bf16)x1.z;  r[7] = (__bf16)x1.w;
  r[8] = (__bf16)y0.x;  r[9] = (__bf16)y0.y;  r[10] = (__bf16)y0.z; r[11] = (__bf16)y0.w;
  r[12] = (__bf16)y1.x; r[13] = (__bf16)y1.y; r[14] = (__bf16)y1.z; r[15] = (__bf16)y1.w;
  return r;
}
__device__ __forceinline__ v16bf cvt_frag_f32m(const float* p0, const float* p1, float msk) {
  float4 x0 = reinterpret_cast<const float4*>(p0)[0];
  float4 x1 = reinterpret_cast<const float4*>(p0)[1];
  float4 y0 = reinterpret_cast<const float4*>(p1)[0];
  float4 y1 = reinterpret_cast<const float4*>(p1)[1];
  v16bf r;
  r[0] = (__bf16)(x0.x * msk);  r[1] = (__bf16)(x0.y * msk);
  r[2] = (__bf16)(x0.z * msk);  r[3] = (__bf16)(x0.w * msk);
  r[4] = (__bf16)(x1.x * msk);  r[5] = (__bf16)(x1.y * msk);
  r[6] = (__bf16)(x1.z * msk);  r[7] = (__bf16)(x1.w * msk);
  r[8] = (__bf16)(y0.x * msk);  r[9] = (__bf16)(y0.y * msk);
  r[10] = (__bf16)(y0.z * msk); r[11] = (__bf16)(y0.w * msk);
  r[12] = (__bf16)(y1.x * msk); r[13] = (__bf16)(y1.y * msk);
  r[14] = (__bf16)(y1.z * msk); r[15] = (__bf16)(y1.w * msk);
  return r;
}
__device__ __forceinline__ v8f wmma_bf16(v16bf a, v16bf b, v8f c) {
  return __builtin_amdgcn_wmma_f32_16x16x32_bf16(false, a, false, b, (short)0, c, false, false);
}
// Per-lane fragment coordinates (contiguous K order used here, wave32):
//   A: lane<16 -> M=lane, K run kb+0..15; lane>=16 -> M=lane-16, K run kb+16..31
//   B: N = lane&15, K run = kb + (lane>>4)*16 .. +15 (two 8-elem halves)
//   D: N = lane&15, M = vgpr + 8*(lane>=16)

// ---------- scalar block layout (floats) ----------
// [0..127] read strengths (bh=b*4+r), [128..159] write strengths, [160..191] cb,
// [192] ca, [193] fg, [194] sum(w), [200..263] e, [264..327] v,
// [328..455] m0, [456..583] m1, [584..711] m2

// =====================================================================
// 1) init
// =====================================================================
__global__ void k_init(float* fwd, float* bwd, float* alloc, float* sc,
                       float* ucopy, const float* __restrict__ usage) {
  int i = blockIdx.x * blockDim.x + threadIdx.x;
  int stride = gridDim.x * blockDim.x;
  for (int idx = i; idx < 128 * KN; idx += stride) { fwd[idx] = 0.f; bwd[idx] = 0.f; }
  for (int idx = i; idx < KN; idx += stride) { alloc[idx] = 0.f; ucopy[idx] = usage[idx]; }
  for (int idx = i; idx < 1024; idx += stride) sc[idx] = 0.f;
}

// =====================================================================
// 2) LSTM gates (WMMA, gatesT[1024][32]); 3 uniform K segments, no branches
// =====================================================================
__global__ void __launch_bounds__(32) k_gates(
    const float* __restrict__ x, const float* __restrict__ pr,
    const float* __restrict__ h0, const float* __restrict__ Wih,
    const float* __restrict__ Whh, const float* __restrict__ bih,
    const float* __restrict__ bhh, float* __restrict__ gatesT) {
  const int jt = blockIdx.x & 63, bt = blockIdx.x >> 6;
  const int j0 = jt * 16, b0 = bt * 16;
  const int lane = threadIdx.x & 31;
  const int m = lane & 15;
  const int klo = (lane < 16) ? 0 : 8;
  const int kb16 = (lane >> 4) * 16;
  v8f acc = {};
  const float* aW = Wih + (size_t)(j0 + m) * 768;
  {  // K 0..511: input_data
    const float* br = x + (size_t)(b0 + m) * 512;
    for (int kb = 0; kb < 512; kb += 32) {
      v16bf a = cvt_frag_f32(aW + kb + kb16, aW + kb + kb16 + 8);
      v16bf b = cvt_frag_f32(br + kb + kb16, br + kb + kb16 + 8);
      acc = wmma_bf16(a, b, acc);
    }
  }
  {  // K 512..767: prev_reads
    const float* ar = aW + 512;
    const float* br = pr + (size_t)(b0 + m) * 256;
    for (int kb = 0; kb < 256; kb += 32) {
      v16bf a = cvt_frag_f32(ar + kb + kb16, ar + kb + kb16 + 8);
      v16bf b = cvt_frag_f32(br + kb + kb16, br + kb + kb16 + 8);
      acc = wmma_bf16(a, b, acc);
    }
  }
  {  // K 768..1023: h0 with W_hh
    const float* ar = Whh + (size_t)(j0 + m) * 256;
    const float* br = h0 + (size_t)(b0 + m) * 256;
    for (int kb = 0; kb < 256; kb += 32) {
      v16bf a = cvt_frag_f32(ar + kb + kb16, ar + kb + kb16 + 8);
      v16bf b = cvt_frag_f32(br + kb + kb16, br + kb + kb16 + 8);
      acc = wmma_bf16(a, b, acc);
    }
  }
  const int col = lane & 15, rb = (lane >= 16) ? 8 : 0;
#pragma unroll
  for (int v = 0; v < 8; ++v) {
    int j = j0 + rb + v, b = b0 + col;
    gatesT[j * 32 + b] = acc[v] + bih[j] + bhh[j];
  }
}

// =====================================================================
// 3) LSTM cell nonlinearity -> h (d_out)
// =====================================================================
__global__ void k_cell(const float* __restrict__ gatesT, const float* __restrict__ c0,
                       float* __restrict__ h) {
  int idx = blockIdx.x * blockDim.x + threadIdx.x;
  if (idx >= KB * KH) return;
  int b = idx & 31, j = idx >> 5;
  float gi = gatesT[(j)*32 + b], gf = gatesT[(j + 256) * 32 + b];
  float gg = gatesT[(j + 512) * 32 + b], go = gatesT[(j + 768) * 32 + b];
  float c = sigf(gf) * c0[b * 256 + j] + sigf(gi) * tanhf(gg);
  h[b * 256 + j] = sigf(go) * tanhf(c);
}

// =====================================================================
// 4) interface GEMM (WMMA) ifaceT[480(468 used)][32]; mask instead of branch
// =====================================================================
__global__ void __launch_bounds__(32) k_iface(
    const float* __restrict__ Wif, const float* __restrict__ bif,
    const float* __restrict__ h, float* __restrict__ ifT) {
  const int it = blockIdx.x % 30, bt = blockIdx.x / 30;
  const int i0 = it * 16, b0 = bt * 16;
  const int lane = threadIdx.x & 31;
  const int m = lane & 15;
  const int klo = (lane < 16) ? 0 : 8;
  const int kb16 = (lane >> 4) * 16;
  const int irow = i0 + m;
  const float msk = (irow < 468) ? 1.0f : 0.0f;
  const float* ar = Wif + (size_t)((irow < 468) ? irow : 467) * 256;
  const float* br = h + (size_t)(b0 + m) * 256;
  v8f acc = {};
  for (int kb = 0; kb < 256; kb += 32) {
    v16bf a = cvt_frag_f32m(ar + kb + kb16, ar + kb + kb16 + 8, msk);
    v16bf b = cvt_frag_f32(br + kb + kb16, br + kb + kb16 + 8);
    acc = wmma_bf16(a, b, acc);
  }
  const int col = lane & 15, rb = (lane >= 16) ? 8 : 0;
#pragma unroll
  for (int v = 0; v < 8; ++v) {
    int i = i0 + rb + v;
    if (i < 468) ifT[i * 32 + b0 + col] = acc[v] + bif[i];
  }
}

// =====================================================================
// 5) parse interface -> scalar block (1 wave, lane=b)
// =====================================================================
__global__ void k_parse(const float* __restrict__ ifT, float* __restrict__ sc) {
  __shared__ float sh[32];
  int b = threadIdx.x;
  for (int r = 0; r < 4; ++r) sc[b * 4 + r] = softplusf(ifT[(256 + r) * 32 + b]) + 1.0f;
  sc[128 + b] = softplusf(ifT[324 * 32 + b]) + 1.0f;
  float wg = sigf(ifT[455 * 32 + b]);
  float ag = sigf(ifT[454 * 32 + b]);
  float fgb = sigf(ifT[453 * 32 + b]);
  sc[160 + b] = wg * (1.0f - ag) * (1.0f / 32.0f);
  for (int r = 0; r < 4; ++r) {
    float a0 = ifT[(456 + r * 3 + 0) * 32 + b];
    float a1 = ifT[(456 + r * 3 + 1) * 32 + b];
    float a2 = ifT[(456 + r * 3 + 2) * 32 + b];
    float mx = fmaxf(a0, fmaxf(a1, a2));
    float e0 = expf(a0 - mx), e1 = expf(a1 - mx), e2 = expf(a2 - mx);
    float s = e0 + e1 + e2;
    sc[328 + b * 4 + r] = e0 / s;
    sc[456 + b * 4 + r] = e1 / s;
    sc[584 + b * 4 + r] = e2 / s;
  }
  sh[b] = wg * ag; __syncthreads();
  if (b == 0) { float s = 0; for (int i = 0; i < 32; ++i) s += sh[i]; sc[192] = s * (1.f / 32.f); }
  __syncthreads();
  sh[b] = fgb; __syncthreads();
  if (b == 0) { float s = 0; for (int i = 0; i < 32; ++i) s += sh[i]; sc[193] = s * (1.f / 32.f); }
  __syncthreads();
  for (int c = 0; c < 64; ++c) {
    sh[b] = sigf(ifT[(389 + c) * 32 + b]); __syncthreads();
    if (b == 0) { float s = 0; for (int i = 0; i < 32; ++i) s += sh[i]; sc[200 + c] = s * (1.f / 32.f); }
    __syncthreads();
    sh[b] = ifT[(325 + c) * 32 + b]; __syncthreads();
    if (b == 0) { float s = 0; for (int i = 0; i < 32; ++i) s += sh[i]; sc[264 + c] = s * (1.f / 32.f); }
    __syncthreads();
  }
}

// =====================================================================
// 6) memory row L2-normalize -> bf16
// =====================================================================
__global__ void k_memnorm(const float* __restrict__ mem, __bf16* __restrict__ mn) {
  __shared__ float sh[256];
  int row = blockIdx.x * 4 + (threadIdx.x >> 6);
  int c = threadIdx.x & 63;
  float v = mem[row * 64 + c];
  sh[threadIdx.x] = v * v;
  __syncthreads();
  for (int off = 32; off >= 1; off >>= 1) {
    if ((threadIdx.x & 63) < off) sh[threadIdx.x] += sh[threadIdx.x + off];
    __syncthreads();
  }
  float inv = 1.0f / fmaxf(sqrtf(sh[threadIdx.x & ~63]), 1e-12f);
  mn[row * 64 + c] = (__bf16)(v * inv);
}

// =====================================================================
// 7) key normalize -> bf16  (rows 0..127 read keys bh=b*4+r, 128..159 write keys)
// =====================================================================
__global__ void k_keynorm(const float* __restrict__ ifT, __bf16* __restrict__ kn) {
  __shared__ float sh[64];
  int row = blockIdx.x, c = threadIdx.x;
  float v;
  if (row < 128) { int b = row >> 2, r = row & 3; v = ifT[(r * 64 + c) * 32 + b]; }
  else { int b = row - 128; v = ifT[(260 + c) * 32 + b]; }
  sh[c] = v * v;
  __syncthreads();
  for (int off = 32; off >= 1; off >>= 1) {
    if (c < off) sh[c] += sh[c + off];
    __syncthreads();
  }
  float inv = 1.0f / fmaxf(sqrtf(sh[0]), 1e-12f);
  kn[row * 64 + c] = (__bf16)(v * inv);
}

// =====================================================================
// 8) cosine similarities (WMMA): cw[bh][n] = keysn[bh,:] . memnorm[n,:]
// =====================================================================
__global__ void __launch_bounds__(32) k_sims(
    const __bf16* __restrict__ mn, const __bf16* __restrict__ kn, float* __restrict__ cw) {
  const int nt = blockIdx.x & 511, bt = blockIdx.x >> 9;
  const int n0 = nt * 16, bh0 = bt * 16;
  const int lane = threadIdx.x & 31;
  const int m = lane & 15;
  const int klo = (lane < 16) ? 0 : 8;
  const int kb16 = (lane >> 4) * 16;
  const __bf16* ar = mn + (size_t)(n0 + m) * 64;
  const __bf16* br = kn + (size_t)(bh0 + m) * 64;
  v8f acc = {};
#pragma unroll
  for (int kb = 0; kb < 64; kb += 32) {
    v16bf a = ld_frag_bf(ar + kb + kb16, ar + kb + kb16 + 8);
    v16bf b = ld_frag_bf(br + kb + kb16, br + kb + kb16 + 8);
    acc = wmma_bf16(a, b, acc);
  }
  const int col = lane & 15, rb = (lane >= 16) ? 8 : 0;
#pragma unroll
  for (int v = 0; v < 8; ++v)
    cw[(size_t)(bh0 + col) * KN + n0 + rb + v] = acc[v];
}

// =====================================================================
// 9) top-10 least used -> allocation weighting (matches argsort[:10])
// =====================================================================
__global__ void k_top10(float* __restrict__ uc, float* __restrict__ alloc) {
  __shared__ float sv[256];
  __shared__ int si[256];
  int t = threadIdx.x;
  for (int it = 0; it < 10; ++it) {
    float bv = 3.0e38f; int bi = 0x7fffffff;
    for (int i = t; i < KN; i += 256) {
      float v = uc[i];
      if (v < bv || (v == bv && i < bi)) { bv = v; bi = i; }
    }
    sv[t] = bv; si[t] = bi; __syncthreads();
    for (int off = 128; off >= 1; off >>= 1) {
      if (t < off && (sv[t + off] < sv[t] || (sv[t + off] == sv[t] && si[t + off] < si[t]))) {
        sv[t] = sv[t + off]; si[t] = si[t + off];
      }
      __syncthreads();
    }
    if (t == 0) {
      int idx = si[0];
      alloc[idx] = ((float)(10 - it) / 10.0f) / (5.5f + 1e-8f);
      uc[idx] = 3.4e38f;
    }
    __syncthreads();
  }
}

// =====================================================================
// 10) sharpened softmax over N per head row; rows<128 also emit bf16 rcw
// =====================================================================
__global__ void k_softmax(float* __restrict__ cw, const float* __restrict__ sc,
                          __bf16* __restrict__ rcwbf) {
  __shared__ float sh[256];
  int row = blockIdx.x, t = threadIdx.x;
  float s = (row < 128) ? sc[row] : sc[128 + (row - 128)];
  float* p = cw + (size_t)row * KN;
  float mx = -3.4e38f;
  for (int i = t; i < KN; i += 256) mx = fmaxf(mx, s * p[i]);
  sh[t] = mx; __syncthreads();
  for (int off = 128; off >= 1; off >>= 1) { if (t < off) sh[t] = fmaxf(sh[t], sh[t + off]); __syncthreads(); }
  mx = sh[0]; __syncthreads();
  float sum = 0.f;
  for (int i = t; i < KN; i += 256) sum += expf(s * p[i] - mx);
  sh[t] = sum; __syncthreads();
  for (int off = 128; off >= 1; off >>= 1) { if (t < off) sh[t] += sh[t + off]; __syncthreads(); }
  float inv = 1.0f / sh[0];
  for (int i = t; i < KN; i += 256) {
    float v = expf(s * p[i] - mx) * inv;
    p[i] = v;
    if (row < 128) rcwbf[(size_t)row * KN + i] = (__bf16)v;
  }
}

// =====================================================================
// 11) batch-mean write weight w[n] and sum(w)
// =====================================================================
__global__ void k_w(const float* __restrict__ cw, const float* __restrict__ alloc,
                    float* __restrict__ sc, float* __restrict__ w) {
  __shared__ float sh[256];
  int n = blockIdx.x * 256 + threadIdx.x;
  float wv = sc[192] * alloc[n];
  for (int b = 0; b < 32; ++b) wv += sc[160 + b] * cw[(size_t)(128 + b) * KN + n];
  w[n] = wv;
  sh[threadIdx.x] = wv; __syncthreads();
  for (int off = 128; off >= 1; off >>= 1) { if (threadIdx.x < off) sh[threadIdx.x] += sh[threadIdx.x + off]; __syncthreads(); }
  if (threadIdx.x == 0) atomicAdd(&sc[194], sh[0]);
}

// =====================================================================
// 12) HEAVY: fused fwd/bwd link products (WMMA bf16) + new_link rewrite.
// 512 column strips x 4 K-segments; each wave holds all 8 r-tiles of both
// accumulators so L is streamed once per orientation. new_link is fused on
// the fwd-orientation loads; rcw A-fragments are raw uint4 loads (L2 hits).
// =====================================================================
__global__ void __launch_bounds__(32) k_link(
    const float* __restrict__ L, const __bf16* __restrict__ rcwbf,
    const float* __restrict__ w, const float* __restrict__ prec,
    float* __restrict__ fwd, float* __restrict__ bwd, float* __restrict__ nl) {
  const int strip = blockIdx.x & 511;
  const int seg = blockIdx.x >> 9;
  const int m0 = strip * 16;
  const int lane = threadIdx.x & 31;
  const int col = lane & 15;
  const int klo = (lane < 16) ? 0 : 8;
  const int kb16 = (lane >> 4) * 16;
  v8f facc[8], bacc[8];
#pragma unroll
  for (int i = 0; i < 8; ++i) { facc[i] = {}; bacc[i] = {}; }
  const float wj = w[m0 + col];
  const float pj = prec[m0 + col];
  const float* Lrow = L + (size_t)(m0 + col) * KN;  // for bwd (transposed) fragment
  const int nend = (seg + 1) * 2048;
  for (int nb = seg * 2048; nb < nend; nb += 32) {
    // fwd B-fragment = L[n-block, m-strip]; new_link fused on the same elements
    v16bf bF;
#pragma unroll
    for (int e = 0; e < 16; ++e) {
      int i = nb + kb16 + e;
      float lv = L[(size_t)i * KN + m0 + col];
      float wi = w[i];
      float out = (1.0f - wi - wj) * lv + wi * pj;
      if (i == m0 + col) out = 0.0f;
      nl[(size_t)i * KN + m0 + col] = out;
      bF[e] = (__bf16)lv;
    }
    // bwd B-fragment = L[m-strip rows, n-block]: contiguous 128b loads + cvt
    v16bf bB = cvt_frag_f32(Lrow + nb + kb16, Lrow + nb + kb16 + 8);
#pragma unroll
    for (int rt = 0; rt < 8; ++rt) {
      const __bf16* ar = rcwbf + (size_t)(rt * 16 + (lane & 15)) * KN + nb;
      v16bf a = ld_frag_bf(ar + kb16, ar + kb16 + 8);
      facc[rt] = wmma_bf16(a, bF, facc[rt]);
      bacc[rt] = wmma_bf16(a, bB, bacc[rt]);
    }
  }
  const int rb = (lane >= 16) ? 8 : 0;
#pragma unroll
  for (int rt = 0; rt < 8; ++rt) {
#pragma unroll
    for (int v = 0; v < 8; ++v) {
      int row = rt * 16 + rb + v;
      atomicAdd(&fwd[(size_t)row * KN + m0 + col], facc[rt][v]);
      atomicAdd(&bwd[(size_t)row * KN + m0 + col], bacc[rt][v]);
    }
  }
}

// =====================================================================
// 13) mean read weight per slot (for usage retention)
// =====================================================================
__global__ void k_mrw(const float* __restrict__ fwd, const float* __restrict__ bwd,
                      const float* __restrict__ cw, const float* __restrict__ sc,
                      float* __restrict__ mrw) {
  int n = blockIdx.x * 256 + threadIdx.x;
  float s = 0.f;
  for (int bh = 0; bh < 128; ++bh)
    s += sc[328 + bh] * bwd[(size_t)bh * KN + n] + sc[456 + bh] * cw[(size_t)bh * KN + n] +
         sc[584 + bh] * fwd[(size_t)bh * KN + n];
  mrw[n] = s * (1.0f / 128.0f);
}

// =====================================================================
// 14) read vectors: rv[bh][w] = sum_n rw(bh,n) * mem[n][w]
// =====================================================================
__global__ void k_rv(const float* __restrict__ fwd, const float* __restrict__ bwd,
                     const float* __restrict__ cw, const float* __restrict__ sc,
                     const float* __restrict__ mem, float* __restrict__ rv) {
  __shared__ float lds[256];
  __shared__ float part[256];
  int bh = blockIdx.x, t = threadIdx.x;
  float m0v = sc[328 + bh], m1v = sc[456 + bh], m2v = sc[584 + bh];
  int wcol = t & 63, p = t >> 6;
  float acc = 0.f;
  for (int nb = 0; nb < KN; nb += 256) {
    int n = nb + t;
    lds[t] = m0v * bwd[(size_t)bh * KN + n] + m1v * cw[(size_t)bh * KN + n] +
             m2v * fwd[(size_t)bh * KN + n];
    __syncthreads();
    int i0 = p * 64;
    for (int i = 0; i < 64; ++i) acc += lds[i0 + i] * mem[(size_t)(nb + i0 + i) * 64 + wcol];
    __syncthreads();
  }
  part[t] = acc; __syncthreads();
  if (t < 64) rv[bh * 64 + t] = part[t] + part[t + 64] + part[t + 128] + part[t + 192];
}

// =====================================================================
// 15) memory erase/write update
// =====================================================================
__global__ void k_newmem(const float* __restrict__ mem, const float* __restrict__ w,
                         const float* __restrict__ sc, float* __restrict__ nm) {
  int idx = blockIdx.x * 256 + threadIdx.x;
  int i = idx >> 6, j = idx & 63;
  float wi = w[i];
  nm[idx] = mem[idx] * (1.0f - wi * sc[200 + j]) + wi * sc[264 + j];
}

// =====================================================================
// 16) usage + precedence updates
// =====================================================================
__global__ void k_up(const float* __restrict__ usage, const float* __restrict__ prec,
                     const float* __restrict__ w, const float* __restrict__ mrw,
                     const float* __restrict__ sc, float* __restrict__ nu, float* __restrict__ np) {
  int n = blockIdx.x * 256 + threadIdx.x;
  float wv = w[n], u = usage[n];
  nu[n] = (u + wv - u * wv) * (1.0f - sc[193] * mrw[n]);
  np[n] = (1.0f - sc[194]) * prec[n] + wv;
}

// =====================================================================
extern "C" void kernel_launch(void* const* d_in, const int* in_sizes, int n_in,
                              void* d_out, int out_size, void* d_ws, size_t ws_size,
                              hipStream_t stream) {
  (void)in_sizes; (void)n_in; (void)out_size; (void)ws_size;
  const float* in_x    = (const float*)d_in[0];   // input_data [32,512]
  const float* in_pr   = (const float*)d_in[1];   // prev_reads [32,256]
  const float* in_h0   = (const float*)d_in[2];   // h0 [32,256]
  const float* in_c0   = (const float*)d_in[3];   // c0 [32,256]
  const float* in_mem  = (const float*)d_in[4];   // memory_matrix [8192,64]
  const float* in_use  = (const float*)d_in[5];   // usage_vector [8192]
  const float* in_link = (const float*)d_in[6];   // link_matrix [8192,8192]
  const float* in_prec = (const float*)d_in[7];   // precedence_weights [8192]
  const float* in_Wih  = (const float*)d_in[8];   // [1024,768]
  const float* in_Whh  = (const float*)d_in[9];   // [1024,256]
  const float* in_bih  = (const float*)d_in[10];  // [1024]
  const float* in_bhh  = (const float*)d_in[11];  // [1024]
  const float* in_Wif  = (const float*)d_in[12];  // [468,256]
  const float* in_bif  = (const float*)d_in[13];  // [468]

  float* out = (float*)d_out;
  float* o_rv    = out;                 // 8192
  float* o_h     = out + 8192;          // 8192
  float* o_mem   = out + 16384;         // 524288
  float* o_usage = out + 540672;        // 8192
  float* o_link  = out + 548864;        // 67108864
  float* o_prec  = out + 67657728;      // 8192

  char* wsb = (char*)d_ws;
  float*  gatesT  = (float*)(wsb + 0);                 // 1024*32 f32
  float*  ifaceT  = (float*)(wsb + 131072);            // 480*32 f32
  __bf16* keysn   = (__bf16*)(wsb + 196608);           // 160*64 bf16
  __bf16* memnorm = (__bf16*)(wsb + 217088);           // 8192*64 bf16
  float*  cw      = (float*)(wsb + 1265664);           // 160*8192 f32
  __bf16* rcwbf   = (__bf16*)(wsb + 6508544);          // 128*8192 bf16
  float*  fwd     = (float*)(wsb + 8605696);           // 128*8192 f32
  float*  bwd     = (float*)(wsb + 12800000);          // 128*8192 f32
  float*  ucopy   = (float*)(wsb + 16994304);          // 8192 f32
  float*  alloc   = (float*)(wsb + 17027072);          // 8192 f32
  float*  wvec    = (float*)(wsb + 17059840);          // 8192 f32
  float*  mrw     = (float*)(wsb + 17092608);          // 8192 f32
  float*  sc      = (float*)(wsb + 17125376);          // 1024 f32

  k_init<<<2048, 256, 0, stream>>>(fwd, bwd, alloc, sc, ucopy, in_use);
  k_gates<<<128, 32, 0, stream>>>(in_x, in_pr, in_h0, in_Wih, in_Whh, in_bih, in_bhh, gatesT);
  k_cell<<<32, 256, 0, stream>>>(gatesT, in_c0, o_h);
  k_iface<<<60, 32, 0, stream>>>(in_Wif, in_bif, o_h, ifaceT);
  k_parse<<<1, 32, 0, stream>>>(ifaceT, sc);
  k_memnorm<<<2048, 256, 0, stream>>>(in_mem, memnorm);
  k_keynorm<<<160, 64, 0, stream>>>(ifaceT, keysn);
  k_sims<<<5120, 32, 0, stream>>>(memnorm, keysn, cw);
  k_top10<<<1, 256, 0, stream>>>(ucopy, alloc);
  k_softmax<<<160, 256, 0, stream>>>(cw, sc, rcwbf);
  k_w<<<32, 256, 0, stream>>>(cw, alloc, sc, wvec);
  k_link<<<2048, 32, 0, stream>>>(in_link, rcwbf, wvec, in_prec, fwd, bwd, o_link);
  k_mrw<<<32, 256, 0, stream>>>(fwd, bwd, cw, sc, mrw);
  k_rv<<<128, 256, 0, stream>>>(fwd, bwd, cw, sc, in_mem, o_rv);
  k_newmem<<<2048, 256, 0, stream>>>(in_mem, wvec, sc, o_mem);
  k_up<<<32, 256, 0, stream>>>(in_use, in_prec, wvec, mrw, sc, o_usage, o_prec);
}
